// MoELayer_50422916055538
// MI455X (gfx1250) — matrix operand order repaired and checked by reading, not compiled, not run on hardware
//
#include <hip/hip_runtime.h>
#include <hip/hip_bf16.h>

// ---------------------------------------------------------------------------
// MoE layer for MI455X (gfx1250): bf16 WMMA GEMMs, fp32 accumulate,
// register-staged double-buffered LDS (1 barrier per K-step).
//   out = shared(x) + sum_e combine[t,e] * (gelu(x@w1_e + b1_e) @ w2_e + b2_e)
// d_out = [final (T*D f32)] ++ [expert_mask (T*2*8 f32)]
// ---------------------------------------------------------------------------

typedef __bf16 bf16_t;
typedef __attribute__((ext_vector_type(16))) __bf16 v16bf;
typedef __attribute__((ext_vector_type(8)))  float  v8f;

union FragBF {
  v16bf v;
  uint4 q[2];
};

// Probe: hardware packed f32->bf16 convert via inline asm (builtin not
// declared in this toolchain). If the assembler rejects v_cvt_pk_bf16_f32,
// set this to 0 to restore the software RNE pack.
#define USE_HW_CVT_PK_BF16 1

#if USE_HW_CVT_PK_BF16
__device__ __forceinline__ unsigned pack_bf2(float a, float b) {
  unsigned r;
  asm("v_cvt_pk_bf16_f32 %0, %1, %2" : "=v"(r) : "v"(a), "v"(b));
  return r;
}
#else
__device__ __forceinline__ unsigned short f2bf_bits_sw(float f) {
  unsigned u = __builtin_bit_cast(unsigned, f);
  unsigned r = u + 0x7FFFu + ((u >> 16) & 1u);   // round-to-nearest-even
  return (unsigned short)(r >> 16);
}
__device__ __forceinline__ unsigned pack_bf2(float a, float b) {
  return (unsigned)f2bf_bits_sw(a) | ((unsigned)f2bf_bits_sw(b) << 16);
}
#endif

__device__ __forceinline__ bf16_t f2bf(float f) {
  const unsigned p = pack_bf2(f, 0.0f);
  return __builtin_bit_cast(bf16_t, (unsigned short)(p & 0xFFFFu));
}
__device__ __forceinline__ float gelu_exact(float v) {
  return 0.5f * v * (1.0f + erff(v * 0.70710678118654752440f));
}

#define BM 128
#define BN 128
#define BK 32
#define PAD_A 40   // 32 + 8 halfword pad (80B row stride -> conflict-free b128)
#define PAD_B 40

// ---------------------------------------------------------------------------
// Router: logits = x @ gate_w [T,8]; top-2 softmax -> combine[T,8], mask[T,2,8]
// One wave32 per token.
// ---------------------------------------------------------------------------
__global__ __launch_bounds__(128) void router_kernel(
    const float* __restrict__ x, const float* __restrict__ gw,
    float* __restrict__ combine, float* __restrict__ mask,
    int T, int D)
{
  const int lane = threadIdx.x & 31;
  const int wid  = threadIdx.x >> 5;
  const int t = blockIdx.x * 4 + wid;
  if (t >= T) return;

  float acc[8] = {0.f,0.f,0.f,0.f,0.f,0.f,0.f,0.f};
  const float* xr = x + (size_t)t * D;
  for (int d = lane; d < D; d += 32) {
    const float xv = xr[d];
    const float4 g0 = *(const float4*)&gw[(size_t)d * 8];
    const float4 g1 = *(const float4*)&gw[(size_t)d * 8 + 4];
    acc[0] += xv * g0.x; acc[1] += xv * g0.y; acc[2] += xv * g0.z; acc[3] += xv * g0.w;
    acc[4] += xv * g1.x; acc[5] += xv * g1.y; acc[6] += xv * g1.z; acc[7] += xv * g1.w;
  }
  #pragma unroll
  for (int e = 0; e < 8; e++) {
    #pragma unroll
    for (int off = 16; off > 0; off >>= 1)
      acc[e] += __shfl_xor(acc[e], off, 32);
  }
  if (lane == 0) {
    int i0 = 0;
    #pragma unroll
    for (int e = 1; e < 8; e++) if (acc[e] > acc[i0]) i0 = e;
    int i1 = (i0 == 0) ? 1 : 0;
    #pragma unroll
    for (int e = 0; e < 8; e++) if (e != i0 && acc[e] > acc[i1]) i1 = e;
    const float e1v = expf(acc[i1] - acc[i0]);   // softmax over [v0, v1], v0 max
    const float s = 1.0f + e1v;
    const float p0 = 1.0f / s, p1 = e1v / s;

    float* cr = combine + (size_t)t * 8;
    #pragma unroll
    for (int e = 0; e < 8; e++) cr[e] = 0.0f;
    cr[i0] = p0; cr[i1] = p1;

    float* mr = mask + (size_t)t * 16;
    #pragma unroll
    for (int e = 0; e < 16; e++) mr[e] = 0.0f;
    mr[i0]     = 1.0f;
    mr[8 + i1] = 1.0f;
  }
}

// ---------------------------------------------------------------------------
// GEMM 1: H = gelu(A[f32 M,K] @ B[f32 K,N] + bias[N]) -> bf16 H[M,N]
// 128x128x32 tiles, 8 waves (2x4), each wave 4x2 WMMA 16x16 tiles.
// Double-buffered LDS, register staging, one barrier per K-step.
// ---------------------------------------------------------------------------
__global__ __launch_bounds__(256) void gemm_x_w1_gelu(
    const float* __restrict__ A, const float* __restrict__ B,
    const float* __restrict__ bias, bf16_t* __restrict__ H,
    int M, int N, int K)
{
  __shared__ bf16_t sA[2][BM * PAD_A];
  __shared__ bf16_t sB[2][BN * PAD_B];   // transposed: [n][k]

  const int tid   = threadIdx.x;
  const int lane  = tid & 31;
  const int wave  = tid >> 5;
  const int waveM = wave >> 2;        // 0..1
  const int waveN = wave & 3;         // 0..3
  const int l15   = lane & 15;
  const int lhalf = lane >> 4;

  const int blockN = blockIdx.x * BN;
  const int blockM = blockIdx.y * BM;

  v8f acc[4][2];
  #pragma unroll
  for (int i = 0; i < 4; i++)
    #pragma unroll
    for (int j = 0; j < 2; j++)
      acc[i][j] = (v8f){0.f,0.f,0.f,0.f,0.f,0.f,0.f,0.f};

  // A staging: thread covers rows aRow+{0,32,64,96}, 4 consecutive K
  const int aRow = tid >> 3;          // 0..31
  const int aK   = (tid & 7) << 2;    // 0..28
  // B staging: thread covers k-pairs (tid>>5)+{0,8} (*2), 4 consecutive N
  const int bKp  = tid >> 5;          // 0..7
  const int bN0  = (tid & 31) << 2;   // 0..124

  float4 ra[4];
  float4 rb[2][2];

  auto load_global = [&](int kt) {
    #pragma unroll
    for (int r = 0; r < 4; r++)
      ra[r] = *(const float4*)&A[(size_t)(blockM + aRow + 32 * r) * K + kt + aK];
    #pragma unroll
    for (int q = 0; q < 2; q++) {
      const int k = (bKp + q * 8) * 2;
      rb[q][0] = *(const float4*)&B[(size_t)(kt + k)     * N + blockN + bN0];
      rb[q][1] = *(const float4*)&B[(size_t)(kt + k + 1) * N + blockN + bN0];
    }
  };
  auto store_lds = [&](int buf) {
    #pragma unroll
    for (int r = 0; r < 4; r++) {
      uint2 p;
      p.x = pack_bf2(ra[r].x, ra[r].y);
      p.y = pack_bf2(ra[r].z, ra[r].w);
      *(uint2*)&sA[buf][(aRow + 32 * r) * PAD_A + aK] = p;
    }
    #pragma unroll
    for (int q = 0; q < 2; q++) {
      const int k = (bKp + q * 8) * 2;
      const float4 f0 = rb[q][0], f1 = rb[q][1];
      *(unsigned*)&sB[buf][(bN0 + 0) * PAD_B + k] = pack_bf2(f0.x, f1.x);
      *(unsigned*)&sB[buf][(bN0 + 1) * PAD_B + k] = pack_bf2(f0.y, f1.y);
      *(unsigned*)&sB[buf][(bN0 + 2) * PAD_B + k] = pack_bf2(f0.z, f1.z);
      *(unsigned*)&sB[buf][(bN0 + 3) * PAD_B + k] = pack_bf2(f0.w, f1.w);
    }
  };
  auto compute = [&](int buf) {
    FragBF fa[4], fb[2];
    // A 16x32 bf16: lanes 0-15 row m hold K 0..7 & 16..23; lanes 16-31 K 8..15 & 24..31
    #pragma unroll
    for (int i = 0; i < 4; i++) {
      const int row  = waveM * 64 + i * 16 + l15;
      const int base = row * PAD_A + lhalf * 8;
      fa[i].q[0] = *(const uint4*)&sA[buf][base];
      fa[i].q[1] = *(const uint4*)&sA[buf][base + 16];
    }
    // B 32x16 bf16: lanes 0-15 col n hold K 0..7 & 16..23; lanes 16-31 K 8..15 & 24..31
    #pragma unroll
    for (int j = 0; j < 2; j++) {
      const int col  = waveN * 32 + j * 16 + l15;
      const int base = col * PAD_B + lhalf * 8;
      fb[j].q[0] = *(const uint4*)&sB[buf][base];
      fb[j].q[1] = *(const uint4*)&sB[buf][base + 16];
    }
    #pragma unroll
    for (int i = 0; i < 4; i++)
      #pragma unroll
      for (int j = 0; j < 2; j++)
        acc[i][j] = __builtin_amdgcn_wmma_f32_16x16x32_bf16(
            false, fa[i].v, false, fb[j].v, (short)0, acc[i][j], false, false);
  };

  load_global(0);
  store_lds(0);
  __syncthreads();

  int cur = 0;
  for (int kt = 0; kt < K; kt += BK) {
    const int nxt = kt + BK;
    const bool more = nxt < K;
    if (more) {
      load_global(nxt);
      if (nxt + BK < K) {   // prefetch two tiles ahead (global_prefetch_b8)
        __builtin_prefetch(&B[(size_t)(nxt + BK + bKp * 2) * N + blockN + bN0], 0, 1);
      }
    }
    compute(cur);
    if (more) {
      store_lds(cur ^ 1);
      __syncthreads();
      cur ^= 1;
    }
  }

  // Epilogue: bias + exact GELU, bf16 out
  #pragma unroll
  for (int i = 0; i < 4; i++) {
    #pragma unroll
    for (int j = 0; j < 2; j++) {
      const int col = blockN + waveN * 32 + j * 16 + l15;
      const float bv = bias[col];
      #pragma unroll
      for (int r = 0; r < 8; r++) {
        const int row = blockM + waveM * 64 + i * 16 + r + lhalf * 8;
        const float v = acc[i][j][r] + bv;
        H[(size_t)row * N + col] = f2bf(gelu_exact(v));
      }
    }
  }
}

// ---------------------------------------------------------------------------
// GEMM 2: Out (+)= rowscale * (Hin[bf16 M,K] @ B[f32 K,N] + bias[N])
// scale == nullptr -> plain store (shared expert); else accumulate scaled.
// ---------------------------------------------------------------------------
__global__ __launch_bounds__(256) void gemm_h_w2_out(
    const bf16_t* __restrict__ Hin, const float* __restrict__ B,
    const float* __restrict__ bias, const float* __restrict__ scale, int expert,
    float* __restrict__ Out, int M, int N, int K)
{
  __shared__ bf16_t sA[2][BM * PAD_A];
  __shared__ bf16_t sB[2][BN * PAD_B];

  const int tid   = threadIdx.x;
  const int lane  = tid & 31;
  const int wave  = tid >> 5;
  const int waveM = wave >> 2;
  const int waveN = wave & 3;
  const int l15   = lane & 15;
  const int lhalf = lane >> 4;

  const int blockN = blockIdx.x * BN;
  const int blockM = blockIdx.y * BM;

  v8f acc[4][2];
  #pragma unroll
  for (int i = 0; i < 4; i++)
    #pragma unroll
    for (int j = 0; j < 2; j++)
      acc[i][j] = (v8f){0.f,0.f,0.f,0.f,0.f,0.f,0.f,0.f};

  const int aRow = tid >> 1;          // 0..127
  const int aK   = (tid & 1) << 4;    // 0 or 16
  const int bKp  = tid >> 5;
  const int bN0  = (tid & 31) << 2;

  uint4  ha[2];
  float4 rb[2][2];

  auto load_global = [&](int kt) {
    const size_t gb = (size_t)(blockM + aRow) * K + kt + aK;
    ha[0] = *(const uint4*)&Hin[gb];
    ha[1] = *(const uint4*)&Hin[gb + 8];
    #pragma unroll
    for (int q = 0; q < 2; q++) {
      const int k = (bKp + q * 8) * 2;
      rb[q][0] = *(const float4*)&B[(size_t)(kt + k)     * N + blockN + bN0];
      rb[q][1] = *(const float4*)&B[(size_t)(kt + k + 1) * N + blockN + bN0];
    }
  };
  auto store_lds = [&](int buf) {
    *(uint4*)&sA[buf][aRow * PAD_A + aK]     = ha[0];
    *(uint4*)&sA[buf][aRow * PAD_A + aK + 8] = ha[1];
    #pragma unroll
    for (int q = 0; q < 2; q++) {
      const int k = (bKp + q * 8) * 2;
      const float4 f0 = rb[q][0], f1 = rb[q][1];
      *(unsigned*)&sB[buf][(bN0 + 0) * PAD_B + k] = pack_bf2(f0.x, f1.x);
      *(unsigned*)&sB[buf][(bN0 + 1) * PAD_B + k] = pack_bf2(f0.y, f1.y);
      *(unsigned*)&sB[buf][(bN0 + 2) * PAD_B + k] = pack_bf2(f0.z, f1.z);
      *(unsigned*)&sB[buf][(bN0 + 3) * PAD_B + k] = pack_bf2(f0.w, f1.w);
    }
  };
  auto compute = [&](int buf) {
    FragBF fa[4], fb[2];
    #pragma unroll
    for (int i = 0; i < 4; i++) {
      const int row  = waveM * 64 + i * 16 + l15;
      const int base = row * PAD_A + lhalf * 8;
      fa[i].q[0] = *(const uint4*)&sA[buf][base];
      fa[i].q[1] = *(const uint4*)&sA[buf][base + 16];
    }
    #pragma unroll
    for (int j = 0; j < 2; j++) {
      const int col  = waveN * 32 + j * 16 + l15;
      const int base = col * PAD_B + lhalf * 8;
      fb[j].q[0] = *(const uint4*)&sB[buf][base];
      fb[j].q[1] = *(const uint4*)&sB[buf][base + 16];
    }
    #pragma unroll
    for (int i = 0; i < 4; i++)
      #pragma unroll
      for (int j = 0; j < 2; j++)
        acc[i][j] = __builtin_amdgcn_wmma_f32_16x16x32_bf16(
            false, fa[i].v, false, fb[j].v, (short)0, acc[i][j], false, false);
  };

  load_global(0);
  store_lds(0);
  __syncthreads();

  int cur = 0;
  for (int kt = 0; kt < K; kt += BK) {
    const int nxt = kt + BK;
    const bool more = nxt < K;
    if (more) {
      load_global(nxt);
      if (nxt + BK < K) {
        __builtin_prefetch(&B[(size_t)(nxt + BK + bKp * 2) * N + blockN + bN0], 0, 1);
      }
    }
    compute(cur);
    if (more) {
      store_lds(cur ^ 1);
      __syncthreads();
      cur ^= 1;
    }
  }

  #pragma unroll
  for (int i = 0; i < 4; i++) {
    #pragma unroll
    for (int j = 0; j < 2; j++) {
      const int col = blockN + waveN * 32 + j * 16 + l15;
      const float bv = bias[col];
      #pragma unroll
      for (int r = 0; r < 8; r++) {
        const int row = blockM + waveM * 64 + i * 16 + r + lhalf * 8;
        float v = acc[i][j][r] + bv;
        const size_t o = (size_t)row * N + col;
        if (scale) {
          v *= scale[(size_t)row * 8 + expert];
          Out[o] += v;
        } else {
          Out[o] = v;
        }
      }
    }
  }
}

// ---------------------------------------------------------------------------
extern "C" void kernel_launch(void* const* d_in, const int* in_sizes, int n_in,
                              void* d_out, int out_size, void* d_ws, size_t ws_size,
                              hipStream_t stream) {
  const float* x   = (const float*)d_in[0];
  const float* gw  = (const float*)d_in[1];
  const float* w1  = (const float*)d_in[2];
  const float* b1  = (const float*)d_in[3];
  const float* w2  = (const float*)d_in[4];
  const float* b2  = (const float*)d_in[5];
  const float* sw1 = (const float*)d_in[6];
  const float* sb1 = (const float*)d_in[7];
  const float* sw2 = (const float*)d_in[8];
  const float* sb2 = (const float*)d_in[9];
  (void)n_in; (void)out_size; (void)ws_size;

  const int E = 8;
  const int D = in_sizes[1] / E;   // 1024
  const int F = in_sizes[3] / E;   // 4096
  const int T = in_sizes[0] / D;   // 4096

  float* out  = (float*)d_out;
  float* mask = out + (size_t)T * D;

  float* combine = (float*)d_ws;
  const size_t comb_bytes = (((size_t)T * E * sizeof(float)) + 255) & ~(size_t)255;
  bf16_t* h = (bf16_t*)((char*)d_ws + comb_bytes);   // [T, F] bf16 scratch

  router_kernel<<<(T + 3) / 4, 128, 0, stream>>>(x, gw, combine, mask, T, D);

  const dim3 blk(256);
  const dim3 g1(F / BN, T / BM);
  const dim3 g2(D / BN, T / BM);

  // Shared expert: writes Out
  gemm_x_w1_gelu<<<g1, blk, 0, stream>>>(x, sw1, sb1, h, T, F, D);
  gemm_h_w2_out<<<g2, blk, 0, stream>>>(h, sw2, sb2, nullptr, 0, out, T, D, F);

  // Routed experts: accumulate combine[t,e] * expert_e(x) into Out
  for (int e = 0; e < E; e++) {
    gemm_x_w1_gelu<<<g1, blk, 0, stream>>>(
        x, w1 + (size_t)e * D * F, b1 + (size_t)e * F, h, T, F, D);
    gemm_h_w2_out<<<g2, blk, 0, stream>>>(
        h, w2 + (size_t)e * F * D, b2 + (size_t)e * D, combine, e, out, T, D, F);
  }
}
